// Attention_6571299963239
// MI455X (gfx1250) — matrix operand order repaired and checked by reading, not compiled, not run on hardware
//
#include <hip/hip_runtime.h>
#include <hip/hip_bf16.h>

// ---------------------------------------------------------------------------
// Problem constants (from reference)
// ---------------------------------------------------------------------------
#define DIMC  2048
#define NH    16
#define HD    128
#define BB    4
#define TT    2048
#define FDIM  (NH * HD)      // 2048
#define QKVN  (3 * FDIM)     // 6144
#define MROWS (BB * TT)      // 8192

typedef __attribute__((ext_vector_type(16))) __bf16 v16bf;
typedef __attribute__((ext_vector_type(8)))  float  v8f;
typedef __attribute__((ext_vector_type(4)))  unsigned int u32x4;
typedef __attribute__((ext_vector_type(2)))  float  f32x2;
typedef __attribute__((ext_vector_type(4)))  int    i32x4;

union F16Frag { v16bf v; u32x4 q[2]; };

// ---------------------------------------------------------------------------
// CDNA5 async global->LDS copy (ASYNCcnt path), with portable fallback.
// Builtin signature (from compiler diagnostic): param0 = int4 addrspace(1)*,
// so pass typed vector pointers in the right address spaces.
// ---------------------------------------------------------------------------
#if __has_builtin(__builtin_amdgcn_global_load_async_to_lds_b128)
#define USE_ASYNC 1
#else
#define USE_ASYNC 0
#endif

#if USE_ASYNC
#if __has_builtin(__builtin_amdgcn_s_wait_asynccnt)
#define WAIT_ASYNC(n) __builtin_amdgcn_s_wait_asynccnt(n)
#else
#define WAIT_ASYNC(n) asm volatile("s_wait_asynccnt " #n ::: "memory")
#endif
typedef __attribute__((address_space(1))) i32x4 gvec_t;   // global int4
typedef __attribute__((address_space(3))) i32x4 lvec_t;   // LDS int4
__device__ __forceinline__ void async_cp16(const unsigned short* g, unsigned short* l) {
    __builtin_amdgcn_global_load_async_to_lds_b128(
        (gvec_t*)(unsigned long long)g,
        (lvec_t*)(unsigned int)(unsigned long long)l,
        0, 0);
}
#else
#define WAIT_ASYNC(n)
__device__ __forceinline__ void async_cp16(const unsigned short* g, unsigned short* l) {
    *(u32x4*)l = *(const u32x4*)g;
}
#endif

__device__ __forceinline__ unsigned short f2bf(float f) {
    unsigned int u = __float_as_uint(f);
    u += 0x7FFFu + ((u >> 16) & 1u);   // round-to-nearest-even
    return (unsigned short)(u >> 16);
}

// ---------------------------------------------------------------------------
// fp32 -> bf16, 2 elements per thread, packed 32-bit stores
// ---------------------------------------------------------------------------
__global__ __launch_bounds__(256)
void convert_bf16(const float* __restrict__ in, unsigned short* __restrict__ out) {
    size_t i = (size_t)blockIdx.x * 256 + threadIdx.x;     // pair index
    f32x2 v = *(const f32x2*)(in + 2 * i);
    ((unsigned int*)out)[i] =
        (unsigned int)f2bf(v.x) | ((unsigned int)f2bf(v.y) << 16);
}

// ---------------------------------------------------------------------------
// fp32 [K,N] -> bf16 [N,K] (transpose).  Reads coalesced along n.
// ---------------------------------------------------------------------------
__global__ __launch_bounds__(256)
void convert_transpose_bf16(const float* __restrict__ W, unsigned short* __restrict__ Wt,
                            int K, int N) {
    size_t idx = (size_t)blockIdx.x * 256 + threadIdx.x;   // k*N + n
    int n = (int)(idx % (size_t)N);
    int k = (int)(idx / (size_t)N);
    Wt[(size_t)n * K + k] = f2bf(W[idx]);
}

// ---------------------------------------------------------------------------
// C[M,N] = A[M,K] x Bt[N,K]^T.  A,Bt bf16; C fp32.
// Block: 256 threads = 8 waves in 4(M) x 2(N).  Block tile 128x128, K-step 64.
// Wave tile: 32(M) x 64(N) = 2 A-frags x 4 B-frags -> 8 accumulators.
// Double-buffered LDS; staging via async global->LDS b128 copies.
// ---------------------------------------------------------------------------
__global__ __launch_bounds__(256)
void gemm_bf16(const unsigned short* __restrict__ A, const unsigned short* __restrict__ Bt,
               float* __restrict__ C, int M, int N, int K) {
    constexpr int AST = 72;                 // LDS row stride (bf16), 144 B = 9 x 16 B
    constexpr int BST = 72;
    __shared__ unsigned short sA[2][128 * AST];   // [buf][m][k]  128 x 64
    __shared__ unsigned short sB[2][128 * BST];   // [buf][n][k]  128 x 64

    const int tid   = threadIdx.x;
    const int wave  = tid >> 5;
    const int lane  = tid & 31;
    const int lhalf = lane >> 4;            // 0: lanes 0-15, 1: lanes 16-31
    const int l16   = lane & 15;
    const int waveM = wave >> 1;            // 0..3  -> rows waveM*32
    const int waveN = wave & 1;             // 0..1  -> cols waveN*64
    const int m0    = blockIdx.x * 128;
    const int n0    = blockIdx.y * 128;

    // stage one 128x64 A tile + 128x64 B tile into buffer b (8 async ops/wave)
    auto stage = [&](int b, int k0) {
        #pragma unroll
        for (int i = 0; i < 4; ++i) {
            int c  = tid + i * 256;
            int r  = c >> 3;
            int kc = (c & 7) * 8;
            async_cp16(A + (size_t)(m0 + r) * K + (k0 + kc), &sA[b][r * AST + kc]);
        }
        #pragma unroll
        for (int i = 0; i < 4; ++i) {
            int c  = tid + i * 256;
            int r  = c >> 3;
            int kc = (c & 7) * 8;
            async_cp16(Bt + (size_t)(n0 + r) * K + (k0 + kc), &sB[b][r * BST + kc]);
        }
    };

    v8f acc[2][4] = {};
    stage(0, 0);
    const int nt = K >> 6;
    for (int it = 0; it < nt; ++it) {
        const int cur = it & 1;
        if (it + 1 < nt) {                    // overlap: fill other buffer
            stage(cur ^ 1, (it + 1) << 6);
            WAIT_ASYNC(8);                    // tile `it` complete; next 8 in flight
        } else {
            WAIT_ASYNC(0);
        }
        __syncthreads();

        #pragma unroll
        for (int kk = 0; kk < 64; kk += 32) {
            // A fragments (16x32): lane l<16 holds K 0-15, l>=16 holds K 16-31
            F16Frag af[2];
            #pragma unroll
            for (int i = 0; i < 2; ++i) {
                const unsigned short* ab =
                    &sA[cur][(waveM * 32 + i * 16 + l16) * AST + kk];
                af[i].q[0] = *(const u32x4*)(ab + (lhalf ? 16 : 0));
                af[i].q[1] = *(const u32x4*)(ab + (lhalf ? 24 : 8));
            }
            // B fragments (32x16): lane l<16 holds K 0-15, l>=16 holds K 16-31
            #pragma unroll
            for (int j = 0; j < 4; ++j) {
                F16Frag bf;
                const unsigned short* bb =
                    &sB[cur][(waveN * 64 + j * 16 + l16) * BST + kk + (lhalf ? 16 : 0)];
                bf.q[0] = *(const u32x4*)(bb);
                bf.q[1] = *(const u32x4*)(bb + 8);
                #pragma unroll
                for (int i = 0; i < 2; ++i)
                    acc[i][j] = __builtin_amdgcn_wmma_f32_16x16x32_bf16(
                        false, af[i].v, false, bf.v, (short)0, acc[i][j], false, false);
            }
        }
        __syncthreads();
    }

    // --- write C: lane holds col l16 (+16j), rows r + 8*lhalf (+16i)
    #pragma unroll
    for (int i = 0; i < 2; ++i) {
        #pragma unroll
        for (int j = 0; j < 4; ++j) {
            int col = n0 + waveN * 64 + j * 16 + l16;
            #pragma unroll
            for (int r = 0; r < 8; ++r) {
                int row = m0 + waveM * 32 + i * 16 + r + lhalf * 8;
                C[(size_t)row * N + col] = acc[i][j][r];
            }
        }
    }
}

// ---------------------------------------------------------------------------
// RoPE + scatter: qkv fp32 [B,T,3,H,D] -> q,k,v bf16 [B,H,T,D].
// ---------------------------------------------------------------------------
__global__ __launch_bounds__(256)
void rope_scatter(const float* __restrict__ qkv,
                  const float* __restrict__ cosb, const float* __restrict__ sinb,
                  unsigned short* __restrict__ qo, unsigned short* __restrict__ ko,
                  unsigned short* __restrict__ vo) {
    size_t idx = (size_t)blockIdx.x * 256 + threadIdx.x;   // (b,t,h,dd) dd in [0,64)
    int dd = (int)(idx & 63);
    size_t r = idx >> 6;
    int h = (int)(r & (NH - 1)); r >>= 4;
    int t = (int)(r & (TT - 1));
    int b = (int)(r >> 11);
    int d0 = dd * 2;

    const size_t base = ((size_t)b * TT + t) * QKVN + (size_t)h * HD;
    f32x2 qv = *(const f32x2*)(qkv + base + 0 * FDIM + d0);
    f32x2 kv = *(const f32x2*)(qkv + base + 1 * FDIM + d0);
    f32x2 vv = *(const f32x2*)(qkv + base + 2 * FDIM + d0);
    f32x2 c  = *(const f32x2*)(cosb + (size_t)t * HD + d0);
    f32x2 s  = *(const f32x2*)(sinb + (size_t)t * HD + d0);
    int  dro = (d0 < 64) ? (d0 + 64) : (d0 - 64);
    float sg = (d0 < 64) ? -1.f : 1.f;
    f32x2 qr = *(const f32x2*)(qkv + base + 0 * FDIM + dro);
    f32x2 kr = *(const f32x2*)(qkv + base + 1 * FDIM + dro);

    f32x2 qo2 = qv * c + sg * qr * s;
    f32x2 ko2 = kv * c + sg * kr * s;

    size_t o2 = ((((size_t)b * NH + h) * TT + t) * HD + d0) >> 1;   // uint index
    ((unsigned int*)qo)[o2] = (unsigned int)f2bf(qo2.x) | ((unsigned int)f2bf(qo2.y) << 16);
    ((unsigned int*)ko)[o2] = (unsigned int)f2bf(ko2.x) | ((unsigned int)f2bf(ko2.y) << 16);
    ((unsigned int*)vo)[o2] = (unsigned int)f2bf(vv.x)  | ((unsigned int)f2bf(vv.y)  << 16);
}

// ---------------------------------------------------------------------------
// Flash attention (causal). Grid: (B*NH, T/64); block 128 threads = 4 waves.
// K tile staged via async global->LDS; V tile transposed through registers.
// Output written bf16 to [B, T, H*D] (feeds the output-projection GEMM).
// ---------------------------------------------------------------------------
__global__ __launch_bounds__(128)
void attn_flash(const unsigned short* __restrict__ Q,
                const unsigned short* __restrict__ K,
                const unsigned short* __restrict__ V,
                unsigned short* __restrict__ O) {
    constexpr int KST = 136;                  // sK row stride (bf16): [kc][d]
    constexpr int VST = 40;                   // sV row stride (bf16): [d][kc]
    constexpr int PST = 40;                   // sP row stride (bf16): [qr][kc]
    __shared__ unsigned short sK[32 * KST];
    __shared__ unsigned short sV[128 * VST];
    __shared__ unsigned short sP[4][16 * PST];

    const int tid   = threadIdx.x;
    const int wave  = tid >> 5;
    const int lane  = tid & 31;
    const int lhalf = lane >> 4;
    const int l16   = lane & 15;
    const int bh    = blockIdx.x;             // b*NH + h
    const int q0b   = blockIdx.y * 64;
    const int q0    = q0b + wave * 16;
    const size_t headoff = (size_t)bh * TT * HD;
    const float scale = 0.08838834764831845f; // 1/sqrt(128)

    // Q fragments: 4 x (16x32 bf16), loaded straight from global (row-major in d)
    F16Frag qf[4];
    {
        const unsigned short* qrow = Q + headoff + (size_t)(q0 + l16) * HD;
        #pragma unroll
        for (int kk = 0; kk < 4; ++kk) {
            qf[kk].q[0] = *(const u32x4*)(qrow + kk * 32 + (lhalf ? 16 : 0));
            qf[kk].q[1] = *(const u32x4*)(qrow + kk * 32 + (lhalf ? 24 : 8));
        }
    }

    v8f o_acc[8] = {};
    float mrow[8], lrow[8];
    #pragma unroll
    for (int r = 0; r < 8; ++r) { mrow[r] = -__builtin_inff(); lrow[r] = 0.f; }

    const int kv_end = q0b + 64;
    for (int kv0 = 0; kv0 < kv_end; kv0 += 32) {
        const unsigned short* Kg = K + headoff + (size_t)kv0 * HD;
        const unsigned short* Vg = V + headoff + (size_t)kv0 * HD;
        // --- K tile [32][128]: async b128 copies (512 chunks, 4 per thread)
        #pragma unroll
        for (int i = 0; i < 4; ++i) {
            int c  = tid + i * 128;
            int kc = c >> 4, d0 = (c & 15) * 8;
            async_cp16(Kg + kc * HD + d0, &sK[kc * KST + d0]);
        }
        // --- V tile transposed [128][32]: vector load + 8-way LDS scatter
        #pragma unroll
        for (int i = 0; i < 4; ++i) {
            int c  = tid + i * 128;
            int kc = c >> 4, d0 = (c & 15) * 8;
            union { u32x4 q; unsigned short u[8]; } t;
            t.q = *(const u32x4*)(Vg + kc * HD + d0);
            #pragma unroll
            for (int e = 0; e < 8; ++e) sV[(d0 + e) * VST + kc] = t.u[e];
        }
        WAIT_ASYNC(0);
        __syncthreads();

        // --- scores: S[16 q][32 kc] as two 16x16 f32 accumulators
        v8f s0 = {}, s1 = {};
        #pragma unroll
        for (int kk = 0; kk < 4; ++kk) {
            F16Frag b0, b1;
            const unsigned short* kb0 = &sK[l16 * KST + kk * 32 + (lhalf ? 16 : 0)];
            const unsigned short* kb1 = &sK[(16 + l16) * KST + kk * 32 + (lhalf ? 16 : 0)];
            b0.q[0] = *(const u32x4*)(kb0);  b0.q[1] = *(const u32x4*)(kb0 + 8);
            b1.q[0] = *(const u32x4*)(kb1);  b1.q[1] = *(const u32x4*)(kb1 + 8);
            s0 = __builtin_amdgcn_wmma_f32_16x16x32_bf16(
                false, qf[kk].v, false, b0.v, (short)0, s0, false, false);
            s1 = __builtin_amdgcn_wmma_f32_16x16x32_bf16(
                false, qf[kk].v, false, b1.v, (short)0, s1, false, false);
        }

        // --- scale + causal mask + online softmax (row lives across 16 lanes)
        #pragma unroll
        for (int r = 0; r < 8; ++r) {
            int qrow = q0 + r + lhalf * 8;
            int c0 = kv0 + l16;
            int c1 = kv0 + 16 + l16;
            float a = s0[r] * scale;
            float b = s1[r] * scale;
            if (c0 > qrow) a = -__builtin_inff();
            if (c1 > qrow) b = -__builtin_inff();
            float mx = fmaxf(a, b);
            #pragma unroll
            for (int m = 1; m < 16; m <<= 1)
                mx = fmaxf(mx, __shfl_xor(mx, m, 32));
            float mnew  = fmaxf(mrow[r], mx);
            float alpha = __expf(mrow[r] - mnew);
            mrow[r] = mnew;
            float p0 = __expf(a - mnew);
            float p1 = __expf(b - mnew);
            s0[r] = p0; s1[r] = p1;
            float ls = p0 + p1;
            #pragma unroll
            for (int m = 1; m < 16; m <<= 1)
                ls += __shfl_xor(ls, m, 32);
            lrow[r] = lrow[r] * alpha + ls;
            #pragma unroll
            for (int oc = 0; oc < 8; ++oc) o_acc[oc][r] *= alpha;
        }

        // --- transpose P (C layout -> A layout) through per-wave LDS tile
        {
            unsigned short* pw = sP[wave];
            #pragma unroll
            for (int r = 0; r < 8; ++r) {
                int pr = r + lhalf * 8;
                pw[pr * PST + l16]      = f2bf(s0[r]);
                pw[pr * PST + 16 + l16] = f2bf(s1[r]);
            }
        }
        {
            F16Frag pf;
            const unsigned short* pr_ = &sP[wave][l16 * PST];
            pf.q[0] = *(const u32x4*)(pr_ + (lhalf ? 16 : 0));
            pf.q[1] = *(const u32x4*)(pr_ + (lhalf ? 24 : 8));
            #pragma unroll
            for (int oc = 0; oc < 8; ++oc) {
                F16Frag vf;
                const unsigned short* vb = &sV[(oc * 16 + l16) * VST + (lhalf ? 16 : 0)];
                vf.q[0] = *(const u32x4*)(vb);
                vf.q[1] = *(const u32x4*)(vb + 8);
                o_acc[oc] = __builtin_amdgcn_wmma_f32_16x16x32_bf16(
                    false, pf.v, false, vf.v, (short)0, o_acc[oc], false, false);
            }
        }
        __syncthreads();
    }

    // --- normalize, convert to bf16, write to [B, T, H*D]
    int b = bh >> 4;
    int h = bh & 15;
    #pragma unroll
    for (int oc = 0; oc < 8; ++oc) {
        int d = oc * 16 + l16;
        #pragma unroll
        for (int r = 0; r < 8; ++r) {
            int t = q0 + r + lhalf * 8;
            O[((size_t)b * TT + t) * FDIM + h * HD + d] = f2bf(o_acc[oc][r] / lrow[r]);
        }
    }
}

// ---------------------------------------------------------------------------
extern "C" void kernel_launch(void* const* d_in, const int* in_sizes, int n_in,
                              void* d_out, int out_size, void* d_ws, size_t ws_size,
                              hipStream_t stream) {
    const float* x        = (const float*)d_in[0];
    const float* rope_cos = (const float*)d_in[1];
    const float* rope_sin = (const float*)d_in[2];
    const float* W_qkv    = (const float*)d_in[3];
    const float* W_out    = (const float*)d_in[4];
    float* out = (float*)d_out;

    char* ws = (char*)d_ws;
    size_t off = 0;
    float* qkv = (float*)(ws + off);                 off += (size_t)MROWS * QKVN * 4;  // 192 MB
    unsigned short* xb    = (unsigned short*)(ws + off); off += (size_t)MROWS * DIMC * 2;
    unsigned short* wqkvt = (unsigned short*)(ws + off); off += (size_t)QKVN * DIMC * 2;
    unsigned short* woutt = (unsigned short*)(ws + off); off += (size_t)DIMC * FDIM * 2;
    size_t head_elems = (size_t)BB * NH * TT * HD;
    unsigned short* qb = (unsigned short*)(ws + off); off += head_elems * 2;
    unsigned short* kb = (unsigned short*)(ws + off); off += head_elems * 2;
    unsigned short* vb = (unsigned short*)(ws + off); off += head_elems * 2;
    unsigned short* attn_out = (unsigned short*)qkv;  // reuse qkv region after rope

    // 0) one-time conversions: x -> bf16, weights -> bf16 transposed [N,K]
    convert_bf16<<<(int)(((size_t)MROWS * DIMC / 2) / 256), 256, 0, stream>>>(x, xb);
    convert_transpose_bf16<<<(int)(((size_t)DIMC * QKVN) / 256), 256, 0, stream>>>(
        W_qkv, wqkvt, DIMC, QKVN);
    convert_transpose_bf16<<<(int)(((size_t)FDIM * DIMC) / 256), 256, 0, stream>>>(
        W_out, woutt, FDIM, DIMC);

    // 1) QKV projection: [8192,2048] x [2048,6144] -> fp32 qkv
    dim3 g1(MROWS / 128, QKVN / 128);
    gemm_bf16<<<g1, 256, 0, stream>>>(xb, wqkvt, qkv, MROWS, QKVN, DIMC);

    // 2) RoPE + layout scatter to bf16 [B,H,T,D]
    rope_scatter<<<(int)((head_elems / 2) / 256), 256, 0, stream>>>(
        qkv, rope_cos, rope_sin, qb, kb, vb);

    // 3) causal flash attention -> bf16 [B,T,H*D] (overwrites qkv region)
    attn_flash<<<dim3(BB * NH, TT / 64), 128, 0, stream>>>(qb, kb, vb, attn_out);

    // 4) output projection: [8192,2048] x [2048,2048] -> fp32 out
    dim3 g2(MROWS / 128, FDIM / 128);
    gemm_bf16<<<g2, 256, 0, stream>>>(attn_out, woutt, out, MROWS, FDIM, DIMC);
}
